// MoELayer_63556926046479
// MI455X (gfx1250) — matrix operand order repaired and checked by reading, not compiled, not run on hardware
//
#include <hip/hip_runtime.h>

// ---------------- problem constants ----------------
constexpr int kT   = 2048;           // B*S tokens
constexpr int kD   = 1024;           // model dim
constexpr int kH   = 16;             // heads
constexpr int kHD  = 64;             // head dim
constexpr int kTE  = 16;             // experts
constexpr int kK   = 2;              // top_k
constexpr int kED  = 512;            // expert dim
constexpr int kDS  = 2048;           // shared expert dim
constexpr float kEPS = 1e-5f;
constexpr int kFS  = 24;             // fragment lane stride in halves (48 B)

typedef __attribute__((ext_vector_type(16))) _Float16 v16h;
typedef __attribute__((ext_vector_type(8)))  float    v8f;
typedef v16h v16h_a16 __attribute__((aligned(16)));   // 32B vector, 16B-aligned loads

__device__ inline v16h lds_frag(const _Float16* p) {
  return *reinterpret_cast<const v16h_a16*>(p);       // -> 2x ds_load_b128
}
__device__ inline unsigned pack_f16x2(float lo, float hi) {
  return (unsigned)__builtin_bit_cast(unsigned short, (_Float16)lo) |
         ((unsigned)__builtin_bit_cast(unsigned short, (_Float16)hi) << 16);
}
// A-fragment half-position for K-pair index p (pairs of adjacent K)
__device__ inline void a_pos(int p, int& laneHalf, int& h0) {
  laneHalf = (p >> 3) & 1;
  h0 = 2 * (p & 7);
}

// ---------------- rmsnorm (one block per row) ----------------
__global__ __launch_bounds__(256)
void rmsnorm_kernel(const float* __restrict__ x, const float* __restrict__ w,
                    float* __restrict__ out) {
  __shared__ float red[256];
  const int row = blockIdx.x;
  const float* xr = x + (long)row * kD;
  float s = 0.f;
  for (int d = threadIdx.x; d < kD; d += 256) { float v = xr[d]; s += v * v; }
  red[threadIdx.x] = s;
  __syncthreads();
  for (int st = 128; st > 0; st >>= 1) {
    if (threadIdx.x < st) red[threadIdx.x] += red[threadIdx.x + st];
    __syncthreads();
  }
  const float inv = rsqrtf(red[0] / (float)kD + kEPS);
  for (int d = threadIdx.x; d < kD; d += 256)
    out[(long)row * kD + d] = xr[d] * inv * w[d];
}

// ---------------- generic WMMA GEMM (fragment-major LDS staging) ----------------
// MODE 0: B is [N,K] row-major. Optional residual addend. Direct rows.
// MODE 1: grouped-forward: rows via rowTok; B = Bstack + z*bStride, [K,N] row-major.
// MODE 2: grouped-output: A row = expanded pos; B [N,K]; atomicAdd * rowScale.
template <int MODE>
__global__ __launch_bounds__(256)
void gemm_wmma(const float* __restrict__ A, const float* __restrict__ Bm,
               float* __restrict__ C, int M, int N, int Kd,
               int lda, int ldb, int ldc,
               const float* __restrict__ addend,
               const int* __restrict__ rowTok,
               const float* __restrict__ rowScale,
               const int* __restrict__ counts,
               const int* __restrict__ offsets,
               long bStride) {
  __shared__ __align__(16) _Float16 sAf[4][32][kFS];  // [mtile][lane][half]
  __shared__ __align__(16) _Float16 sBf[4][32][kFS];  // [ntile][lane][half]
  const int tid  = threadIdx.x;
  const int lane = tid & 31;
  const int wave = tid >> 5;
  const int mt   = wave & 3;            // m sub-tile
  const int ncol = (wave >> 2) * 32;    // n offset (2 n sub-tiles per wave)

  int  Mloc = M;
  int  base = 0;
  const float* Bp = Bm;
  if constexpr (MODE == 1 || MODE == 2) {
    const int e = blockIdx.z;
    Mloc = counts[e];
    base = offsets[e];
    Bp   = Bm + (long)e * bStride;
  }
  const int m0 = blockIdx.y * 64;
  if (m0 >= Mloc) return;
  const int n0 = blockIdx.x * 64;

  v8f acc0 = {}; v8f acc1 = {};

  for (int k0 = 0; k0 < Kd; k0 += 32) {
    // ---- stage A tile (64x32) straight into A-fragment layout ----
    #pragma unroll
    for (int i = 0; i < 4; ++i) {
      const int p2 = tid + i * 256;        // K-pair id: 64 rows x 16 pairs
      const int m  = p2 >> 4;
      const int pp = p2 & 15;
      const int kp = pp * 2;
      const int ml = m0 + m;
      float v0 = 0.f, v1 = 0.f;
      if (ml < Mloc) {
        long gr;
        if constexpr (MODE == 1)      gr = rowTok[base + ml];
        else if constexpr (MODE == 2) gr = base + ml;
        else                          gr = ml;
        const float* src = A + gr * (long)lda + (k0 + kp);
        v0 = src[0]; v1 = src[1];
      }
      int lh, h0; a_pos(pp, lh, h0);
      *reinterpret_cast<unsigned*>(&sAf[m >> 4][(m & 15) + 16 * lh][h0]) =
          pack_f16x2(v0, v1);
    }
    // ---- stage B tile (32x64) straight into B-fragment layout ----
    #pragma unroll
    for (int i = 0; i < 4; ++i) {
      const int p2 = tid + i * 256;        // 64 n x 16 k-pairs
      const int n  = p2 >> 4;
      const int kp = (p2 & 15) * 2;
      float v0, v1;
      if constexpr (MODE == 1) {           // [K,N] row-major
        v0 = Bp[(long)(k0 + kp) * ldb + (n0 + n)];
        v1 = Bp[(long)(k0 + kp + 1) * ldb + (n0 + n)];
      } else {                             // [N,K] row-major
        const float* src = Bp + (long)(n0 + n) * ldb + (k0 + kp);
        v0 = src[0]; v1 = src[1];
      }
      *reinterpret_cast<unsigned*>(
          &sBf[n >> 4][(n & 15) + 16 * (kp >> 4)][kp & 15]) = pack_f16x2(v0, v1);
    }
    __syncthreads();

    const v16h a  = lds_frag(&sAf[mt][lane][0]);
    const v16h b0 = lds_frag(&sBf[(ncol >> 4) + 0][lane][0]);
    const v16h b1 = lds_frag(&sBf[(ncol >> 4) + 1][lane][0]);
    acc0 = __builtin_amdgcn_wmma_f32_16x16x32_f16(false, a, false, b0, (short)0, acc0, false, false);
    acc1 = __builtin_amdgcn_wmma_f32_16x16x32_f16(false, a, false, b1, (short)0, acc1, false, false);
    __syncthreads();
  }

  // epilogue: C layout — VGPR r: lanes 0-15 M=r, lanes 16-31 M=r+8
  #pragma unroll
  for (int r = 0; r < 8; ++r) {
    const int ml = m0 + mt * 16 + r + ((lane < 16) ? 0 : 8);
    if (ml >= Mloc) continue;
    const int c0 = n0 + ncol + (lane & 15);
    const float v0 = acc0[r], v1 = acc1[r];
    if constexpr (MODE == 0) {
      const long o0 = (long)ml * ldc + c0;
      C[o0]      = v0 + (addend ? addend[o0]      : 0.f);
      C[o0 + 16] = v1 + (addend ? addend[o0 + 16] : 0.f);
    } else if constexpr (MODE == 1) {
      const long o = (long)(base + ml) * ldc + c0;
      C[o] = v0; C[o + 16] = v1;
    } else {
      const int   tok = rowTok[base + ml];
      const float sc  = rowScale[base + ml];
      const long  o   = (long)tok * ldc + c0;
      atomicAdd(&C[o],      v0 * sc);
      atomicAdd(&C[o + 16], v1 * sc);
    }
  }
}

// ---------------- RoPE + head split: qkv[S,3D] -> q/k/v [H,S,HD] ----------------
__global__ __launch_bounds__(32)
void rope_split_kernel(const float* __restrict__ qkv,
                       float* __restrict__ q, float* __restrict__ k,
                       float* __restrict__ v) {
  const int s = blockIdx.x, h = blockIdx.y, j = threadIdx.x;  // j in [0,32)
  const long base = (long)s * (3 * kD) + h * kHD;
  const float inv = __expf(-((float)(2 * j) / (float)kHD) * 9.210340372f); // ln(1e4)
  const float fr = (float)s * inv;
  const float c = cosf(fr), sn = sinf(fr);
  const long ob = ((long)h * kT + s) * kHD;

  float q1 = qkv[base + j],          q2 = qkv[base + j + 32];
  float k1 = qkv[base + kD + j],     k2 = qkv[base + kD + j + 32];
  q[ob + j]      =  q1 * c + q2 * sn;
  q[ob + j + 32] = -q1 * sn + q2 * c;
  k[ob + j]      =  k1 * c + k2 * sn;
  k[ob + j + 32] = -k1 * sn + k2 * c;
  v[ob + j]      = qkv[base + 2 * kD + j];
  v[ob + j + 32] = qkv[base + 2 * kD + j + 32];
}

// ---------------- flash attention: 4 waves, 64 q-rows per block ----------------
__global__ __launch_bounds__(128)
void flash_attn_kernel(const float* __restrict__ q, const float* __restrict__ k,
                       const float* __restrict__ v, float* __restrict__ out) {
  // K tile as B-fragments for Q*K^T: [step][keytile][lane][half]
  __shared__ __align__(16) _Float16 sKf[2][4][32][kFS];
  // V tile as B-fragments for P*V:   [step][dtile][lane][half]
  __shared__ __align__(16) _Float16 sVf[2][4][32][kFS];
  __shared__ float sP[4][16][64];                       // scores / probs (fp32)
  // P as A-fragments: [wave][step][lane][half]
  __shared__ __align__(16) _Float16 sPf[4][2][32][kFS];
  __shared__ float sStat[4][16][3];                     // m, l, alpha
  const int tid  = threadIdx.x;
  const int lane = tid & 31;
  const int wave = tid >> 5;
  const int h    = blockIdx.y;
  const int q0   = blockIdx.x * 64;
  const float scale = 0.125f;                           // 1/sqrt(64)
  const float* qh = q + (long)h * kT * kHD;
  const float* kh = k + (long)h * kT * kHD;
  const float* vh = v + (long)h * kT * kHD;

  // Q fragments for this wave's 16 rows (once per block; direct global loads)
  v16h aq[2];
  {
    const long mrow = q0 + wave * 16 + (lane & 15);
    #pragma unroll
    for (int s2 = 0; s2 < 2; ++s2)
      #pragma unroll
      for (int pp = 0; pp < 16; ++pp) {
        int lh, h0; a_pos(pp, lh, h0);
        if (lh == ((lane < 16) ? 0 : 1)) {
          const long src = mrow * kHD + 2 * pp + 32 * s2;
          aq[s2][h0]     = (_Float16)(qh[src]     * scale);
          aq[s2][h0 + 1] = (_Float16)(qh[src + 1] * scale);
        }
      }
  }
  v8f o0 = {}, o1 = {}, o2 = {}, o3 = {};
  if (lane < 16) { sStat[wave][lane][0] = -3.0e38f; sStat[wave][lane][1] = 0.f; }

  const int ktMax = blockIdx.x;                         // causal tiles
  for (int kt = 0; kt <= ktMax; ++kt) {
    __syncthreads();
    // ---- stage K tile into QK^T B-fragment layout (pairs along d) ----
    #pragma unroll
    for (int i = 0; i < 16; ++i) {
      const int p2  = tid + i * 128;                    // 64 keys x 32 d-pairs
      const int key = p2 >> 5;
      const int dp  = (p2 & 31) * 2;
      const float* src = kh + (long)(kt * 64 + key) * kHD + dp;
      const int step = dp >> 5, kk = dp & 31;
      *reinterpret_cast<unsigned*>(
          &sKf[step][key >> 4][(key & 15) + 16 * (kk >> 4)][kk & 15]) =
          pack_f16x2(src[0], src[1]);
    }
    // ---- stage V tile into P*V B-fragment layout (pairs along key) ----
    #pragma unroll
    for (int i = 0; i < 16; ++i) {
      const int p2  = tid + i * 128;                    // 64 d x 32 key-pairs
      const int d   = p2 & 63;
      const int keyp = (p2 >> 6) * 2;
      const float* src = vh + (long)(kt * 64 + keyp) * kHD + d;
      const int step = keyp >> 5, kk = keyp & 31;
      *reinterpret_cast<unsigned*>(
          &sVf[step][d >> 4][(d & 15) + 16 * (kk >> 4)][kk & 15]) =
          pack_f16x2(src[0], src[kHD]);
    }
    __syncthreads();

    // S = Q K^T
    v8f s0 = {}, s1 = {}, s2 = {}, s3 = {};
    #pragma unroll
    for (int step = 0; step < 2; ++step) {
      const v16h b0 = lds_frag(&sKf[step][0][lane][0]);
      const v16h b1 = lds_frag(&sKf[step][1][lane][0]);
      const v16h b2 = lds_frag(&sKf[step][2][lane][0]);
      const v16h b3 = lds_frag(&sKf[step][3][lane][0]);
      s0 = __builtin_amdgcn_wmma_f32_16x16x32_f16(false, aq[step], false, b0, (short)0, s0, false, false);
      s1 = __builtin_amdgcn_wmma_f32_16x16x32_f16(false, aq[step], false, b1, (short)0, s1, false, false);
      s2 = __builtin_amdgcn_wmma_f32_16x16x32_f16(false, aq[step], false, b2, (short)0, s2, false, false);
      s3 = __builtin_amdgcn_wmma_f32_16x16x32_f16(false, aq[step], false, b3, (short)0, s3, false, false);
    }
    // dump scores (C layout) to fp32 LDS
    #pragma unroll
    for (int r = 0; r < 8; ++r) {
      const int ml = r + ((lane < 16) ? 0 : 8);
      const int c  = lane & 15;
      sP[wave][ml][c]      = s0[r];
      sP[wave][ml][c + 16] = s1[r];
      sP[wave][ml][c + 32] = s2[r];
      sP[wave][ml][c + 48] = s3[r];
    }
    __syncthreads();
    // online softmax, one row per lane (lanes 0-15)
    if (lane < 16) {
      const int row  = lane;
      const int qidx = q0 + wave * 16 + row;
      const float m  = sStat[wave][row][0];
      const float l  = sStat[wave][row][1];
      float mt = -3.0e38f;
      for (int c = 0; c < 64; ++c) {
        const int kp = kt * 64 + c;
        const float sv = (kp <= qidx) ? sP[wave][row][c] : -3.0e38f;
        mt = fmaxf(mt, sv);
      }
      const float mnew = fmaxf(m, mt);
      float sum = 0.f;
      for (int c = 0; c < 64; ++c) {
        const int kp = kt * 64 + c;
        const float p = (kp <= qidx) ? __expf(sP[wave][row][c] - mnew) : 0.f;
        sP[wave][row][c] = p;
        sum += p;
      }
      const float alpha = __expf(m - mnew);
      sStat[wave][row][0] = mnew;
      sStat[wave][row][1] = l * alpha + sum;
      sStat[wave][row][2] = alpha;
    }
    __syncthreads();
    // convert probs -> A-fragment layout (packed pairs, whole wave)
    #pragma unroll
    for (int j = 0; j < 16; ++j) {
      const int row = j;
      const int c   = 2 * lane;                         // even col pair
      const float p0 = sP[wave][row][c];
      const float p1 = sP[wave][row][c + 1];
      const int step = c >> 5, cc = c & 31;
      int lh, h0; a_pos(cc >> 1, lh, h0);
      *reinterpret_cast<unsigned*>(&sPf[wave][step][row + 16 * lh][h0]) =
          pack_f16x2(p0, p1);
    }
    // rescale running O by alpha
    #pragma unroll
    for (int r = 0; r < 8; ++r) {
      const int ml = r + ((lane < 16) ? 0 : 8);
      const float al = sStat[wave][ml][2];
      o0[r] *= al; o1[r] *= al; o2[r] *= al; o3[r] *= al;
    }
    // O += P V
    #pragma unroll
    for (int step = 0; step < 2; ++step) {
      const v16h ap = lds_frag(&sPf[wave][step][lane][0]);
      const v16h b0 = lds_frag(&sVf[step][0][lane][0]);
      const v16h b1 = lds_frag(&sVf[step][1][lane][0]);
      const v16h b2 = lds_frag(&sVf[step][2][lane][0]);
      const v16h b3 = lds_frag(&sVf[step][3][lane][0]);
      o0 = __builtin_amdgcn_wmma_f32_16x16x32_f16(false, ap, false, b0, (short)0, o0, false, false);
      o1 = __builtin_amdgcn_wmma_f32_16x16x32_f16(false, ap, false, b1, (short)0, o1, false, false);
      o2 = __builtin_amdgcn_wmma_f32_16x16x32_f16(false, ap, false, b2, (short)0, o2, false, false);
      o3 = __builtin_amdgcn_wmma_f32_16x16x32_f16(false, ap, false, b3, (short)0, o3, false, false);
    }
  }
  // epilogue: divide by l, write [S, D] with head offset
  #pragma unroll
  for (int r = 0; r < 8; ++r) {
    const int ml = r + ((lane < 16) ? 0 : 8);
    const float l = sStat[wave][ml][1];
    const float inv = (l > 0.f) ? 1.f / l : 0.f;
    const int row = q0 + wave * 16 + ml;
    const int c = lane & 15;
    const long base = (long)row * kD + h * kHD;
    out[base + c]      = o0[r] * inv;
    out[base + c + 16] = o1[r] * inv;
    out[base + c + 32] = o2[r] * inv;
    out[base + c + 48] = o3[r] * inv;
  }
}

// ---------------- router: logits + top-k softmax scores ----------------
__global__ __launch_bounds__(256)
void router_kernel(const float* __restrict__ xffn, const float* __restrict__ keys,
                   const float* __restrict__ bias, const float* __restrict__ values,
                   const int* __restrict__ indices, float* __restrict__ scores) {
  __shared__ float red[kTE][256];
  const int t = blockIdx.x, tid = threadIdx.x;
  float acc[kTE];
  #pragma unroll
  for (int e = 0; e < kTE; ++e) acc[e] = 0.f;
  for (int d = tid; d < kD; d += 256) {
    const float xv = xffn[(long)t * kD + d];
    #pragma unroll
    for (int e = 0; e < kTE; ++e) acc[e] += xv * keys[(long)d * kTE + e];
  }
  #pragma unroll
  for (int e = 0; e < kTE; ++e) red[e][tid] = acc[e];
  __syncthreads();
  for (int st = 128; st > 0; st >>= 1) {
    if (tid < st)
      #pragma unroll
      for (int e = 0; e < kTE; ++e) red[e][tid] += red[e][tid + st];
    __syncthreads();
  }
  if (tid == 0) {
    float vals[kK];
    float m = -3.0e38f;
    for (int j = 0; j < kK; ++j) {
      const int idx = indices[t * kK + j];
      vals[j] = values[t * kK + j] + red[idx][0] + bias[idx];
      m = fmaxf(m, vals[j]);
    }
    float s = 0.f;
    for (int j = 0; j < kK; ++j) { vals[j] = __expf(vals[j] - m); s += vals[j]; }
    for (int j = 0; j < kK; ++j) scores[t * kK + j] = vals[j] / s;  // RSF = 1
  }
}

__global__ void count_kernel(const int* __restrict__ indices, int* __restrict__ counts) {
  const int i = blockIdx.x * blockDim.x + threadIdx.x;
  if (i < kT * kK) atomicAdd(&counts[indices[i]], 1);
}

__global__ void scan_kernel(const int* __restrict__ counts, int* __restrict__ offsets,
                            int* __restrict__ cursor) {
  if (threadIdx.x == 0) {
    int run = 0;
    for (int e = 0; e < kTE; ++e) { offsets[e] = run; run += counts[e]; cursor[e] = 0; }
  }
}

__global__ void fill_rows_kernel(const int* __restrict__ indices,
                                 const float* __restrict__ scores,
                                 const int* __restrict__ offsets,
                                 int* __restrict__ cursor,
                                 int* __restrict__ rowTok,
                                 float* __restrict__ rowScale) {
  const int i = blockIdx.x * blockDim.x + threadIdx.x;
  if (i >= kT * kK) return;
  const int e = indices[i];
  const int slot = atomicAdd(&cursor[e], 1);
  const int pos = offsets[e] + slot;
  rowTok[pos] = i / kK;
  rowScale[pos] = scores[i];
}

// ---------------- elementwise ----------------
__global__ void silu_mul_pair_kernel(const float* __restrict__ u, float* __restrict__ g,
                                     int rows, int cols) {
  const long i = (long)blockIdx.x * blockDim.x + threadIdx.x;
  if (i >= (long)rows * cols) return;
  const long r = i / cols, c = i % cols;
  const float x1 = u[r * (2L * cols) + c];
  const float x2 = u[r * (2L * cols) + cols + c];
  g[i] = (x1 / (1.f + __expf(-x1))) * x2;
}

__global__ void silu_mul_kernel(float* __restrict__ h1, const float* __restrict__ h2, long n) {
  const long i = (long)blockIdx.x * blockDim.x + threadIdx.x;
  if (i >= n) return;
  const float x1 = h1[i];
  h1[i] = (x1 / (1.f + __expf(-x1))) * h2[i];
}

__global__ void final_combine_kernel(const float* __restrict__ y,
                                     const float* __restrict__ coeff,
                                     const float* __restrict__ yshared,
                                     const float* __restrict__ xffnin,
                                     float* __restrict__ out) {
  const long i = (long)blockIdx.x * blockDim.x + threadIdx.x;
  if (i >= (long)kT * kD) return;
  out[i] = y[i] * coeff[i % kD] + yshared[i] + xffnin[i];
}

// ---------------- host launcher ----------------
extern "C" void kernel_launch(void* const* d_in, const int* in_sizes, int n_in,
                              void* d_out, int out_size, void* d_ws, size_t ws_size,
                              hipStream_t stream) {
  const float* x_input     = (const float*)d_in[0];
  const int*   indices     = (const int*)  d_in[1];
  const float* values      = (const float*)d_in[2];
  const float* attn_w      = (const float*)d_in[3];
  const float* attn_o_w    = (const float*)d_in[4];
  const float* attn_norm_w = (const float*)d_in[5];
  const float* ffn_norm_w  = (const float*)d_in[6];
  const float* ffn_experts = (const float*)d_in[7];   // [3, TE, D, ED]
  const float* main_keys   = (const float*)d_in[8];
  const float* main_bias   = (const float*)d_in[9];
  const float* out_coeff   = (const float*)d_in[10];
  const float* ffn_up_w    = (const float*)d_in[11];
  const float* ffn_down_w  = (const float*)d_in[12];
  const float* shared_norm_w = (const float*)d_in[13];
  float* out = (float*)d_out;

  const long TD = (long)kT * kD;
  float* ws    = (float*)d_ws;
  float* xnorm = ws;               float* qkv   = xnorm + TD;
  float* qr    = qkv + 3 * TD;     float* kr    = qr + TD;
  float* vr    = kr + TD;          float* xattn = vr + TD;
  float* xffnin= xattn + TD;       float* xffn  = xffnin + TD;
  float* u     = xffn + TD;        float* gs    = u + 4 * TD;
  float* ysh   = gs + 2 * TD;      float* h1    = ysh + TD;
  float* h2    = h1 + TD;          float* ybuf  = h2 + TD;
  float* scores  = ybuf + TD;
  float* rowScale= scores + kT * kK;
  int* rowTok  = (int*)(rowScale + kT * kK);
  int* counts  = rowTok + kT * kK;
  int* offs    = counts + kTE;
  int* cursor  = offs + kTE;

  const long eW = (long)kTE * kD * kED;
  const long eStride = (long)kD * kED;

  rmsnorm_kernel<<<kT, 256, 0, stream>>>(x_input, attn_norm_w, xnorm);
  gemm_wmma<0><<<dim3(3 * kD / 64, kT / 64, 1), 256, 0, stream>>>(
      xnorm, attn_w, qkv, kT, 3 * kD, kD, kD, kD, 3 * kD,
      nullptr, nullptr, nullptr, nullptr, nullptr, 0);
  rope_split_kernel<<<dim3(kT, kH), 32, 0, stream>>>(qkv, qr, kr, vr);
  flash_attn_kernel<<<dim3(kT / 64, kH), 128, 0, stream>>>(qr, kr, vr, xattn);
  gemm_wmma<0><<<dim3(kD / 64, kT / 64, 1), 256, 0, stream>>>(
      xattn, attn_o_w, xffnin, kT, kD, kD, kD, kD, kD,
      x_input, nullptr, nullptr, nullptr, nullptr, 0);
  rmsnorm_kernel<<<kT, 256, 0, stream>>>(xffnin, ffn_norm_w, xffn);
  router_kernel<<<kT, 256, 0, stream>>>(xffn, main_keys, main_bias, values, indices, scores);
  hipMemsetAsync(counts, 0, kTE * sizeof(int), stream);
  hipMemsetAsync(ybuf, 0, TD * sizeof(float), stream);
  count_kernel<<<(kT * kK + 255) / 256, 256, 0, stream>>>(indices, counts);
  scan_kernel<<<1, 32, 0, stream>>>(counts, offs, cursor);
  fill_rows_kernel<<<(kT * kK + 255) / 256, 256, 0, stream>>>(
      indices, scores, offs, cursor, rowTok, rowScale);
  gemm_wmma<1><<<dim3(kED / 64, (kT * kK) / 64, kTE), 256, 0, stream>>>(
      xffn, ffn_experts + 0 * eW, h1, 0, kED, kD, kD, kED, kED,
      nullptr, rowTok, nullptr, counts, offs, eStride);
  gemm_wmma<1><<<dim3(kED / 64, (kT * kK) / 64, kTE), 256, 0, stream>>>(
      xffn, ffn_experts + 1 * eW, h2, 0, kED, kD, kD, kED, kED,
      nullptr, rowTok, nullptr, counts, offs, eStride);
  silu_mul_kernel<<<(int)(((long)kT * kK * kED + 255) / 256), 256, 0, stream>>>(
      h1, h2, (long)kT * kK * kED);
  gemm_wmma<2><<<dim3(kD / 64, (kT * kK) / 64, kTE), 256, 0, stream>>>(
      h1, ffn_experts + 2 * eW, ybuf, 0, kD, kED, kED, kED, kD,
      nullptr, rowTok, rowScale, counts, offs, eStride);
  gemm_wmma<0><<<dim3(2 * kDS / 64, kT / 64, 1), 256, 0, stream>>>(
      xffn, ffn_up_w, u, kT, 2 * kDS, kD, kD, kD, 2 * kDS,
      nullptr, nullptr, nullptr, nullptr, nullptr, 0);
  silu_mul_pair_kernel<<<(int)(((long)kT * kDS + 255) / 256), 256, 0, stream>>>(
      u, gs, kT, kDS);
  gemm_wmma<0><<<dim3(kD / 64, kT / 64, 1), 256, 0, stream>>>(
      gs, ffn_down_w, ysh, kT, kD, kDS, kDS, kDS, kD,
      nullptr, nullptr, nullptr, nullptr, nullptr, 0);
  rmsnorm_kernel<<<kT, 256, 0, stream>>>(ysh, shared_norm_w, ysh);
  final_combine_kernel<<<(int)((TD + 255) / 256), 256, 0, stream>>>(
      ybuf, out_coeff, ysh, xffnin, out);
}
